// MonotonicAttention_80805514707115
// MI455X (gfx1250) — matrix operand order repaired and checked by reading, not compiled, not run on hardware
//
#include <hip/hip_runtime.h>
#include <hip/hip_bf16.h>
#include <math.h>

// ---------------------------------------------------------------------------
// Monotonic infinite-lookback attention for MI455X (gfx1250, wave32, WMMA,
// async global->LDS staging on the projection GEMMs)
// ---------------------------------------------------------------------------
typedef __attribute__((ext_vector_type(16))) __bf16 v16bf;
typedef __attribute__((ext_vector_type(8)))  __bf16 v8bf;
typedef __attribute__((ext_vector_type(8)))  float  v8f;

#define T_DIM   1024
#define S_DIM   1024
#define B_DIM   4
#define E_DIM   512
#define H_DIM   8
#define HD_DIM  64
#define BH_DIM  32
#define SCALE_F 0.125f          // HD^-0.5
#define EPS_F   1e-6f
#define CUMEPS_F 1e-10f
#define SROW    520             // padded LDS row stride (elements) -> bank spread

// ------------------------------ wave helpers -------------------------------
__device__ __forceinline__ float wave_incl_scan(float x, int lane) {
#pragma unroll
  for (int d = 1; d < 32; d <<= 1) {
    float n = __shfl_up(x, d, 32);
    if (lane >= d) x += n;
  }
  return x;
}
__device__ __forceinline__ float wave_max(float x) {
#pragma unroll
  for (int d = 16; d >= 1; d >>= 1) x = fmaxf(x, __shfl_xor(x, d, 32));
  return x;
}
__device__ __forceinline__ float wave_sum(float x) {
#pragma unroll
  for (int d = 16; d >= 1; d >>= 1) x += __shfl_xor(x, d, 32);
  return x;
}

// ---------------------- async global -> LDS (gfx1250) ----------------------
// GV mode: LDS address is taken from the VDST VGPR; low 32 bits of a generic LDS
// pointer ARE the DS byte offset (aperture rule, ISA 10.2).
__device__ __forceinline__ void async_load_b128(unsigned lds_off, const void* g) {
  asm volatile("global_load_async_to_lds_b128 %0, %1, off"
               :: "v"(lds_off), "v"((unsigned long long)(uintptr_t)g)
               : "memory");
}
__device__ __forceinline__ void wait_async0() {
  asm volatile("s_wait_asynccnt 0" ::: "memory");
}

// --------------------------- WMMA fragment loads ---------------------------
// A fragment: 16x32 bf16 row-major tile. ISA: lanes0-15 row=lane,
// K = {kb..kb+7, kb+16..kb+23}, kb = 8*(lane>>4).
__device__ __forceinline__ v16bf load_a_frag(const __bf16* tile, int ldk, int lane) {
  int row = lane & 15;
  int kb  = (lane >> 4) << 3;
  const __bf16* p = tile + (size_t)row * ldk + kb;
  v8bf lo = *(const v8bf*)(p);
  v8bf hi = *(const v8bf*)(p + 16);
  return __builtin_shufflevector(lo, hi, 0,1,2,3,4,5,6,7,8,9,10,11,12,13,14,15);
}
// B fragment: 32x16 bf16 from an [N,K]-major operand (lane = column n).
// K = {kb..kb+7, kb+16..kb+23}, kb = 8*(lane>>4) (matches A fragment).
__device__ __forceinline__ v16bf load_b_frag(const __bf16* tileNK, int ldk, int lane) {
  int col = lane & 15;
  int kb  = (lane >> 4) << 3;
  const __bf16* p = tileNK + (size_t)col * ldk + kb;
  v8bf lo = *(const v8bf*)(p);
  v8bf hi = *(const v8bf*)(p + 16);
  return __builtin_shufflevector(lo, hi, 0,1,2,3,4,5,6,7,8,9,10,11,12,13,14,15);
}

// ------------------------------ cvt f32->bf16 ------------------------------
__global__ void cvt_f32_bf16(const float* __restrict__ in, __bf16* __restrict__ out, int n) {
  int i = blockIdx.x * 256 + threadIdx.x;
  if (i < n) out[i] = (__bf16)in[i];
}

// ------------------- projection GEMM (WMMA, LDS-staged A) ------------------
// Y[m,n] = (sum_k A[m,k] * W[n,k] + bias[n]) * scale,  M x 512, K = 512
// One block = 256 threads = 8 waves; block owns one 16-row tile of A, staged
// once into LDS via async loads; wave w computes the 64-wide N supertile w.
// MODE 0: head-major bf16 [BH,L,HD]; MODE 1: head-transposed bf16 [BH,HD,L];
// MODE 2: flat f32 [M,512].
template <int MODE>
__global__ void wmma_proj(const __bf16* __restrict__ A, const __bf16* __restrict__ W,
                          const float* __restrict__ bias, float scale,
                          __bf16* __restrict__ outB, float* __restrict__ outF) {
  const int Kd = E_DIM;
  __shared__ __bf16 sA[16 * SROW];
  int tid  = threadIdx.x;
  int lane = tid & 31;
  int tn   = tid >> 5;            // wave id = 64-col supertile
  int tm   = blockIdx.x;          // 16-row tile

  // stage contiguous 16x512 bf16 A tile (16 KB) into padded LDS rows
  {
    const char* gsrc = (const char*)(A + (size_t)tm * 16 * Kd);
    unsigned sbase = (unsigned)(uintptr_t)(void*)sA;
#pragma unroll
    for (int i = 0; i < 4; ++i) {
      int chunk = tid + 256 * i;          // 1024 x 16B chunks
      int row = chunk >> 6;               // 64 chunks per row
      int c16 = chunk & 63;
      async_load_b128(sbase + (unsigned)(row * SROW * 2 + c16 * 16),
                      gsrc + row * 1024 + c16 * 16);
    }
    wait_async0();
    __syncthreads();
  }

  v8f z = {0.f,0.f,0.f,0.f,0.f,0.f,0.f,0.f};
  v8f acc[4] = {z, z, z, z};
  for (int kk = 0; kk < Kd; kk += 32) {
    v16bf af = load_a_frag(sA + kk, SROW, lane);        // ds_load path
#pragma unroll
    for (int j = 0; j < 4; ++j) {
      v16bf bf_ = load_b_frag(W + (size_t)(tn * 64 + j * 16) * Kd + kk, Kd, lane);
      acc[j] = __builtin_amdgcn_wmma_f32_16x16x32_bf16(
          false, af, false, bf_, (short)0, acc[j], false, false);
    }
  }
  int nl = lane & 15;
  int mo = (lane >> 4) << 3;
#pragma unroll
  for (int j = 0; j < 4; ++j) {
#pragma unroll
    for (int r = 0; r < 8; ++r) {
      int m = tm * 16 + mo + r;           // flat row  = l*B + b
      int n = tn * 64 + j * 16 + nl;      // out feature = h*64 + d
      float v = (acc[j][r] + bias[n]) * scale;
      if (MODE == 0) {
        int l = m >> 2, b = m & 3, h = n >> 6, d = n & 63;
        outB[(((size_t)(b * H_DIM + h)) * T_DIM + l) * HD_DIM + d] = (__bf16)v;
      } else if (MODE == 1) {
        int l = m >> 2, b = m & 3, h = n >> 6, d = n & 63;
        outB[(((size_t)(b * H_DIM + h)) * HD_DIM + d) * S_DIM + l] = (__bf16)v;
      } else {
        outF[(size_t)m * E_DIM + n] = v;
      }
    }
  }
}

// -------------------------- energy GEMM (WMMA) -----------------------------
// e[bh,t,s] = sum_d Q[bh,t,d]*K[bh,s,d] + ebias ; optional sigmoid
__global__ void wmma_energy(const __bf16* __restrict__ Q, const __bf16* __restrict__ Kh,
                            const float* __restrict__ ebias, float* __restrict__ out,
                            int do_sigmoid) {
  int lane = threadIdx.x & 31;
  int tt = blockIdx.x;            // t tile (16)
  int ts = blockIdx.y;            // s supertile (64)
  int bh = blockIdx.z;
  const __bf16* q = Q  + (size_t)bh * T_DIM * HD_DIM;
  const __bf16* k = Kh + (size_t)bh * S_DIM * HD_DIM;
  v8f z = {0.f,0.f,0.f,0.f,0.f,0.f,0.f,0.f};
  v8f acc[4] = {z, z, z, z};
#pragma unroll
  for (int kk = 0; kk < HD_DIM; kk += 32) {
    v16bf af = load_a_frag(q + (size_t)tt * 16 * HD_DIM + kk, HD_DIM, lane);
#pragma unroll
    for (int j = 0; j < 4; ++j) {
      v16bf bf_ = load_b_frag(k + (size_t)(ts * 64 + j * 16) * HD_DIM + kk, HD_DIM, lane);
      acc[j] = __builtin_amdgcn_wmma_f32_16x16x32_bf16(
          false, af, false, bf_, (short)0, acc[j], false, false);
    }
  }
  float eb = ebias[0];
  int nl = lane & 15;
  int mo = (lane >> 4) << 3;
#pragma unroll
  for (int j = 0; j < 4; ++j) {
#pragma unroll
    for (int r = 0; r < 8; ++r) {
      int t = tt * 16 + mo + r;
      int s = ts * 64 + j * 16 + nl;
      float e = acc[j][r] + eb;
      if (do_sigmoid) e = 1.0f / (1.0f + expf(-e));
      out[((size_t)bh * T_DIM + t) * S_DIM + s] = e;
    }
  }
}

// ----------------------- per-row prep (one wave/row) -----------------------
// Lane owns 4 consecutive elements; 8 chunks of 128 cover S=1024.
// c = exclusive cumprod(1-p) via logs; soft -> ex (in place) and cs.
__global__ void row_prep(const float* __restrict__ p, float* __restrict__ c,
                         float* __restrict__ soft_ex, float* __restrict__ cs) {
  int row  = blockIdx.x * 8 + (threadIdx.x >> 5);     // (bh*T + t)
  int lane = threadIdx.x & 31;
  size_t base = (size_t)row * S_DIM;
  // ---- exclusive safe-cumprod of (1-p) ----
  float carry = 0.0f;
  for (int ch = 0; ch < 8; ++ch) {
    size_t i = base + ch * 128 + lane * 4;
    float4 pv = *(const float4*)(p + i);
    float lg0 = logf(1.0f - pv.x + CUMEPS_F);
    float lg1 = logf(1.0f - pv.y + CUMEPS_F);
    float lg2 = logf(1.0f - pv.z + CUMEPS_F);
    float lg3 = logf(1.0f - pv.w + CUMEPS_F);
    float tot = lg0 + lg1 + lg2 + lg3;
    float incl = wave_incl_scan(tot, lane);
    float e0 = carry + incl - tot;                     // exclusive base
    float4 cv;
    cv.x = expf(e0);
    cv.y = expf(e0 + lg0);
    cv.z = expf(e0 + lg0 + lg1);
    cv.w = expf(e0 + lg0 + lg1 + lg2);
    *(float4*)(c + i) = cv;
    carry += __shfl(incl, 31, 32);
  }
  // ---- running softmax pieces on soft energies ----
  float m = -1e30f;
  for (int ch = 0; ch < 8; ++ch) {
    float4 sv = *(const float4*)(soft_ex + base + ch * 128 + lane * 4);
    m = fmaxf(m, fmaxf(fmaxf(sv.x, sv.y), fmaxf(sv.z, sv.w)));
  }
  m = wave_max(m);
  carry = 0.0f;
  for (int ch = 0; ch < 8; ++ch) {
    size_t i = base + ch * 128 + lane * 4;
    float4 sv = *(const float4*)(soft_ex + i);
    float4 ev;
    ev.x = expf(sv.x - m); ev.y = expf(sv.y - m);
    ev.z = expf(sv.z - m); ev.w = expf(sv.w - m);
    float tot = ev.x + ev.y + ev.z + ev.w;
    float incl = wave_incl_scan(tot, lane);
    float e0 = carry + incl - tot;
    float4 cv;
    cv.x = e0 + ev.x;
    cv.y = e0 + ev.x + ev.y;
    cv.z = e0 + ev.x + ev.y + ev.z;
    cv.w = e0 + tot;
    *(float4*)(cs + i) = cv;                           // inclusive cumsum
    *(float4*)(soft_ex + i) = ev;                      // overwrite with ex
    carry += __shfl(incl, 31, 32);
  }
}

// --------------------- sequential alpha scan over T ------------------------
// 256 threads x 4 elems; thread-local prefix -> wave scan -> 8-partial LDS.
__global__ void alpha_scan(const float* __restrict__ p, const float* __restrict__ c,
                           float* __restrict__ alpha) {
  int bh = blockIdx.x;
  int tid = threadIdx.x;                 // 0..255
  int lane = tid & 31, wave = tid >> 5;  // 8 waves
  __shared__ float partial[8];
  size_t base = (size_t)bh * T_DIM * S_DIM + (size_t)tid * 4;
  float4 prev;
  prev.x = (tid == 0) ? 1.0f : 0.0f;
  prev.y = prev.z = prev.w = 0.0f;
  for (int t = 0; t < T_DIM; ++t) {
    size_t i = base + (size_t)t * S_DIM;
    float4 cv = *(const float4*)(c + i);
    float4 pv = *(const float4*)(p + i);
    float y0 = prev.x / fminf(fmaxf(cv.x, EPS_F), 1.0f);
    float y1 = prev.y / fminf(fmaxf(cv.y, EPS_F), 1.0f);
    float y2 = prev.z / fminf(fmaxf(cv.z, EPS_F), 1.0f);
    float y3 = prev.w / fminf(fmaxf(cv.w, EPS_F), 1.0f);
    float l0 = y0, l1 = y0 + y1, l2 = l1 + y2, l3 = l2 + y3;
    float incl = wave_incl_scan(l3, lane);
    if (lane == 31) partial[wave] = incl;
    __syncthreads();
    float off = incl - l3;                 // thread-exclusive within wave
#pragma unroll
    for (int w = 0; w < 7; ++w)
      if (w < wave) off += partial[w];     // broadcast LDS reads
    __syncthreads();                        // protect partial for next t
    float4 a;
    a.x = fminf(fmaxf(pv.x * cv.x * (off + l0), 0.0f), 1.0f);
    a.y = fminf(fmaxf(pv.y * cv.y * (off + l1), 0.0f), 1.0f);
    a.z = fminf(fmaxf(pv.z * cv.z * (off + l2), 0.0f), 1.0f);
    a.w = fminf(fmaxf(pv.w * cv.w * (off + l3), 0.0f), 1.0f);
    *(float4*)(alpha + i) = a;
    prev = a;
  }
}

// ---------------- beta = ex * reverse_cumsum(alpha/cs) ---------------------
// suffix[s] = total - exclusive_prefix[s]
__global__ void beta_kernel(const float* __restrict__ alpha, const float* __restrict__ cs,
                            const float* __restrict__ ex, __bf16* __restrict__ beta) {
  int row  = blockIdx.x * 8 + (threadIdx.x >> 5);
  int lane = threadIdx.x & 31;
  size_t base = (size_t)row * S_DIM;
  float tot = 0.0f;
  for (int ch = 0; ch < 8; ++ch) {
    size_t i = base + ch * 128 + lane * 4;
    float4 av = *(const float4*)(alpha + i);
    float4 sv = *(const float4*)(cs + i);
    tot += av.x / sv.x + av.y / sv.y + av.z / sv.z + av.w / sv.w;
  }
  tot = wave_sum(tot);
  float carry = 0.0f;
  for (int ch = 0; ch < 8; ++ch) {
    size_t i = base + ch * 128 + lane * 4;
    float4 av = *(const float4*)(alpha + i);
    float4 sv = *(const float4*)(cs + i);
    float4 evx = *(const float4*)(ex + i);
    float n0 = av.x / sv.x, n1 = av.y / sv.y, n2 = av.z / sv.z, n3 = av.w / sv.w;
    float ttot = n0 + n1 + n2 + n3;
    float incl = wave_incl_scan(ttot, lane);
    float e0 = carry + incl - ttot;                    // element-exclusive base
    beta[i + 0] = (__bf16)(evx.x * (tot - e0));
    beta[i + 1] = (__bf16)(evx.y * (tot - (e0 + n0)));
    beta[i + 2] = (__bf16)(evx.z * (tot - (e0 + n0 + n1)));
    beta[i + 3] = (__bf16)(evx.w * (tot - (e0 + n0 + n1 + n2)));
    carry += __shfl(incl, 31, 32);
  }
}

// ------------------------ beta x V GEMM (WMMA) -----------------------------
// attn[bh,t,d] = sum_s beta[bh,t,s] * V[bh,s,d];  VT stored [BH,HD,S]
__global__ void wmma_attnv(const __bf16* __restrict__ Beta, const __bf16* __restrict__ VT,
                           __bf16* __restrict__ attnF) {
  int lane = threadIdx.x & 31;
  int tt = blockIdx.x;
  int bh = blockIdx.z;
  const __bf16* brow = Beta + (size_t)bh * T_DIM * S_DIM;
  const __bf16* vt   = VT   + (size_t)bh * HD_DIM * S_DIM;
  v8f z = {0.f,0.f,0.f,0.f,0.f,0.f,0.f,0.f};
  v8f acc[4] = {z, z, z, z};
  const __bf16* abase = brow + (size_t)tt * 16 * S_DIM;
  for (int kk = 0; kk < S_DIM; kk += 32) {
    v16bf af = load_a_frag(abase + kk, S_DIM, lane);
    __builtin_prefetch(abase + kk + 64, 0, 1);
#pragma unroll
    for (int j = 0; j < 4; ++j) {
      v16bf bf_ = load_b_frag(vt + (size_t)(j * 16) * S_DIM + kk, S_DIM, lane);
      acc[j] = __builtin_amdgcn_wmma_f32_16x16x32_bf16(
          false, af, false, bf_, (short)0, acc[j], false, false);
    }
  }
  int b = bh >> 3, h = bh & 7;
  int nl = lane & 15;
  int mo = (lane >> 4) << 3;
#pragma unroll
  for (int j = 0; j < 4; ++j) {
#pragma unroll
    for (int r = 0; r < 8; ++r) {
      int t = tt * 16 + mo + r;
      int d = j * 16 + nl;
      attnF[((size_t)(t * B_DIM + b)) * E_DIM + h * HD_DIM + d] = (__bf16)acc[j][r];
    }
  }
}

// ---------------------------------------------------------------------------
extern "C" void kernel_launch(void* const* d_in, const int* in_sizes, int n_in,
                              void* d_out, int out_size, void* d_ws, size_t ws_size,
                              hipStream_t stream) {
  const float* query = (const float*)d_in[0];
  const float* key   = (const float*)d_in[1];
  const float* value = (const float*)d_in[2];
  const float* qm_w  = (const float*)d_in[3];  const float* qm_b = (const float*)d_in[4];
  const float* km_w  = (const float*)d_in[5];  const float* km_b = (const float*)d_in[6];
  const float* qs_w  = (const float*)d_in[7];  const float* qs_b = (const float*)d_in[8];
  const float* ks_w  = (const float*)d_in[9];  const float* ks_b = (const float*)d_in[10];
  const float* v_w   = (const float*)d_in[11]; const float* v_b  = (const float*)d_in[12];
  const float* out_w = (const float*)d_in[13]; const float* out_b= (const float*)d_in[14];
  const float* ebias = (const float*)d_in[15];
  float* out = (float*)d_out;

  // ---- workspace carve-out (256B aligned regions) ----
  char* w = (char*)d_ws;
  size_t off = 0;
  auto carve = [&](size_t bytes) -> char* {
    char* p = w + off;
    off = (off + bytes + 255) & ~(size_t)255;
    return p;
  };
  const size_t MxE   = (size_t)T_DIM * B_DIM * E_DIM;      // 2,097,152
  const size_t ExE   = (size_t)E_DIM * E_DIM;              //   262,144
  const size_t HEADS = (size_t)BH_DIM * T_DIM * HD_DIM;    // 2,097,152
  const size_t BTS   = (size_t)BH_DIM * T_DIM * S_DIM;     // 33,554,432

  __bf16* qb   = (__bf16*)carve(MxE * 2);
  __bf16* kb   = (__bf16*)carve(MxE * 2);
  __bf16* vb   = (__bf16*)carve(MxE * 2);
  __bf16* wqm  = (__bf16*)carve(ExE * 2);
  __bf16* wkm  = (__bf16*)carve(ExE * 2);
  __bf16* wqs  = (__bf16*)carve(ExE * 2);
  __bf16* wks  = (__bf16*)carve(ExE * 2);
  __bf16* wv   = (__bf16*)carve(ExE * 2);
  __bf16* wout = (__bf16*)carve(ExE * 2);
  __bf16* qmh  = (__bf16*)carve(HEADS * 2);
  __bf16* kmh  = (__bf16*)carve(HEADS * 2);
  __bf16* qsh  = (__bf16*)carve(HEADS * 2);
  __bf16* ksh  = (__bf16*)carve(HEADS * 2);
  __bf16* vth  = (__bf16*)carve(HEADS * 2);
  __bf16* attnF= (__bf16*)carve(MxE * 2);
  __bf16* betaB= (__bf16*)carve(BTS * 2);
  float* pF    = (float*)carve(BTS * 4);
  float* cF    = (float*)carve(BTS * 4);
  float* exF   = (float*)carve(BTS * 4);   // soft energies, then ex in place
  float* csF   = (float*)carve(BTS * 4);
  float* alphaF= (float*)carve(BTS * 4);
  (void)ws_size;

  // 1) convert activations + weights to bf16
  cvt_f32_bf16<<<(int)(MxE / 256), 256, 0, stream>>>(query, qb, (int)MxE);
  cvt_f32_bf16<<<(int)(MxE / 256), 256, 0, stream>>>(key,   kb, (int)MxE);
  cvt_f32_bf16<<<(int)(MxE / 256), 256, 0, stream>>>(value, vb, (int)MxE);
  cvt_f32_bf16<<<(int)(ExE / 256), 256, 0, stream>>>(qm_w, wqm, (int)ExE);
  cvt_f32_bf16<<<(int)(ExE / 256), 256, 0, stream>>>(km_w, wkm, (int)ExE);
  cvt_f32_bf16<<<(int)(ExE / 256), 256, 0, stream>>>(qs_w, wqs, (int)ExE);
  cvt_f32_bf16<<<(int)(ExE / 256), 256, 0, stream>>>(ks_w, wks, (int)ExE);
  cvt_f32_bf16<<<(int)(ExE / 256), 256, 0, stream>>>(v_w,  wv,  (int)ExE);
  cvt_f32_bf16<<<(int)(ExE / 256), 256, 0, stream>>>(out_w,wout,(int)ExE);

  // 2) projections (WMMA bf16, async-LDS staged A; block = 8 waves = full N)
  int pg = T_DIM * B_DIM / 16;   // 256 blocks
  wmma_proj<0><<<pg, 256, 0, stream>>>(qb, wqm, qm_b, SCALE_F, qmh, nullptr);
  wmma_proj<0><<<pg, 256, 0, stream>>>(kb, wkm, km_b, 1.0f,    kmh, nullptr);
  wmma_proj<0><<<pg, 256, 0, stream>>>(qb, wqs, qs_b, SCALE_F, qsh, nullptr);
  wmma_proj<0><<<pg, 256, 0, stream>>>(kb, wks, ks_b, 1.0f,    ksh, nullptr);
  wmma_proj<1><<<pg, 256, 0, stream>>>(vb, wv,  v_b,  1.0f,    vth, nullptr);

  // 3) energies: p = sigmoid(q_m k_m^T + eb); soft raw
  dim3 eg(T_DIM / 16, S_DIM / 64, BH_DIM);
  wmma_energy<<<eg, 32, 0, stream>>>(qmh, kmh, ebias, pF,  1);
  wmma_energy<<<eg, 32, 0, stream>>>(qsh, ksh, ebias, exF, 0);

  // 4) per-row prep: exclusive cumprod(1-p), softmax ex + cumsum
  row_prep<<<BH_DIM * T_DIM / 8, 256, 0, stream>>>(pF, cF, exF, csF);

  // 5) sequential expected-alignment recurrence over T (latency-critical)
  alpha_scan<<<BH_DIM, 256, 0, stream>>>(pF, cF, alphaF);

  // 6) beta = ex * reverse_cumsum(alpha / cs)
  beta_kernel<<<BH_DIM * T_DIM / 8, 256, 0, stream>>>(alphaF, csF, exF, betaB);

  // 7) attn = beta x V  (WMMA), head-interleaved bf16 [T*B, E]
  dim3 ag(T_DIM / 16, 1, BH_DIM);
  wmma_attnv<<<ag, 32, 0, stream>>>(betaB, vth, attnF);

  // 8) output projection (WMMA) + bias -> f32 d_out [T,B,E]
  wmma_proj<2><<<pg, 256, 0, stream>>>(attnF, wout, out_b, 1.0f, nullptr, out);
}
